// Attention_28750511080014
// MI455X (gfx1250) — matrix operand order repaired and checked by reading, not compiled, not run on hardware
//
#include <hip/hip_runtime.h>

typedef __attribute__((ext_vector_type(16))) _Float16 v16h;
typedef __attribute__((ext_vector_type(8)))  _Float16 v8h;
typedef __attribute__((ext_vector_type(8)))  float    v8f;
typedef __attribute__((ext_vector_type(4)))  float    v4f;

constexpr int B_   = 2;
constexpr int S_   = 2048;
constexpr int H_   = 1024;
constexpr int NQ   = 16;
constexpr int HD   = 64;           // head dim
constexpr int GHD  = 256;          // HD * NKV
constexpr int QKVW = H_ + 2 * GHD; // 1536
constexpr int MROWS = B_ * S_;     // 4096 token rows

union AFrag {
    v16h v;
    v8h  h[2];
    _Float16 e[16];
};

// ---------------------------------------------------------------------------
// f32 -> f16 elementwise conversion (grid-stride)
// ---------------------------------------------------------------------------
__global__ void cvt_f32_f16(const float* __restrict__ in,
                            _Float16* __restrict__ out, int n) {
    int i = blockIdx.x * blockDim.x + threadIdx.x;
    int stride = gridDim.x * blockDim.x;
    for (; i < n; i += stride) out[i] = (_Float16)in[i];
}

// ---------------------------------------------------------------------------
// C[M,N] = A[M,K] (f16 row-major) @ W[N,K]^T (f16 row-major)
// One wave owns a 16(M) x 64(N) strip.  K is a compile-time constant so the
// K-loop fully unrolls; fragments are double-buffered: next K-step's loads
// are issued before the current step's 4 v_wmma_f32_16x16x32_f16.
// ---------------------------------------------------------------------------
template <typename OutT, int K>
__global__ void gemm_wmma_f16(const _Float16* __restrict__ A,
                              const _Float16* __restrict__ W,
                              OutT* __restrict__ C, int N) {
    const int lane = threadIdx.x & 31;
    const int hi   = lane >> 4;
    const int l15  = lane & 15;
    const int wave = blockIdx.x * (blockDim.x >> 5) + (threadIdx.x >> 5);
    const int nstrips = N >> 6;
    const int m0 = (wave / nstrips) * 16;
    const int n0 = (wave % nstrips) * 64;

    // A fragment K-map: lanes 0-15 -> {0..15}, lanes 16-31 -> {16..31}
    const _Float16* arow = A + (size_t)(m0 + l15) * K + 16 * hi;
    // B fragment K-map: lanes 0-15 -> {0..15}, lanes 16-31 -> {16..31}
    const _Float16* wr0  = W + (size_t)(n0 + l15) * K + 16 * hi;

    v8f acc[4] = {};
    AFrag a[2];
    AFrag bm[2][4];

    a[0].h[0] = *(const v8h*)(arow);
    a[0].h[1] = *(const v8h*)(arow + 8);
#pragma unroll
    for (int t = 0; t < 4; ++t) {
        const _Float16* w = wr0 + (size_t)t * 16 * K;
        bm[0][t].h[0] = *(const v8h*)(w);
        bm[0][t].h[1] = *(const v8h*)(w + 8);
    }

#pragma unroll
    for (int k0 = 0; k0 < K; k0 += 32) {
        const int cur = (k0 >> 5) & 1;
        const int nxt = cur ^ 1;
        if (k0 + 32 < K) {
            const int kn = k0 + 32;
            a[nxt].h[0] = *(const v8h*)(arow + kn);
            a[nxt].h[1] = *(const v8h*)(arow + kn + 8);
#pragma unroll
            for (int t = 0; t < 4; ++t) {
                const _Float16* w = wr0 + (size_t)t * 16 * K + kn;
                bm[nxt][t].h[0] = *(const v8h*)(w);
                bm[nxt][t].h[1] = *(const v8h*)(w + 8);
            }
        }
#pragma unroll
        for (int t = 0; t < 4; ++t)
            acc[t] = __builtin_amdgcn_wmma_f32_16x16x32_f16(
                false, a[cur].v, false, bm[cur][t].v, (short)0, acc[t],
                false, false);
    }

#pragma unroll
    for (int t = 0; t < 4; ++t)
#pragma unroll
        for (int r = 0; r < 8; ++r)
            C[(size_t)(m0 + r + 8 * hi) * N + n0 + 16 * t + l15] =
                (OutT)acc[t][r];
}

// ---------------------------------------------------------------------------
// Flash attention over the reference's reshape semantics.
// qkv (f16) laid out [B*S, 1536]; with t = head*S + s:
//   q(b,h,s,:)  = qkv[(b*S + t/16)*1536 + (t%16)*64 ...]          (t = h*S+s)
//   k(b,hk,s,:) = qkv[(b*S + t/4 )*1536 + 1024 + (t%4)*64 ...]    (t = hk*S+s)
//   v(b,hk,s,:) = qkv[(b*S + t/4 )*1536 + 1280 + (t%4)*64 ...]
// One wave per (b, h, 16-query tile); 32-key blocks; online softmax in the
// transposed domain (lane = query row, one shfl_xor(16) per reduction).
// K tiles are double-buffered in LDS via GLOBAL_LOAD_ASYNC_TO_LDS_B128
// (ASYNCcnt) with the next block's copy in flight during the current block.
// ---------------------------------------------------------------------------
__global__ void attn_flash_wmma(const _Float16* __restrict__ qkv,
                                _Float16* __restrict__ yt) {
    const int lane = threadIdx.x & 31;
    const int hi   = lane >> 4;
    const int l15  = lane & 15;
    const int q0 = blockIdx.x * 16;
    const int h  = blockIdx.y;
    const int b  = blockIdx.z;
    const int hk = h >> 2;   // h / (NQ/NKV)

    __shared__ _Float16 kS[2][32][72];        // async-staged K tiles [key][d]
    __shared__ _Float16 vT[HD][40];           // transposed V tile [d][key]
    __shared__ float    sS[16][36];           // scores [row][key]
    __shared__ _Float16 pT[16][40];           // probabilities [row][key]
    __shared__ alignas(16) float bcast[16];   // per-row alpha / 1/l broadcast

    // ---- async copy of one K tile (32 keys x 64 f16 = 32 x 128B rows) ----
    auto issueK = [&](int jblk, int buf) {
        const int tk = hk * S_ + (jblk + lane);
        const _Float16* gk =
            qkv + (size_t)(b * S_ + (tk >> 2)) * QKVW + H_ + (tk & 3) * HD;
        const unsigned la = (unsigned)(size_t)&kS[buf][lane][0];
        asm volatile(
            "global_load_async_to_lds_b128 %0, %1, off offset:0\n\t"
            "global_load_async_to_lds_b128 %0, %1, off offset:16\n\t"
            "global_load_async_to_lds_b128 %0, %1, off offset:32\n\t"
            "global_load_async_to_lds_b128 %0, %1, off offset:48\n\t"
            "global_load_async_to_lds_b128 %0, %1, off offset:64\n\t"
            "global_load_async_to_lds_b128 %0, %1, off offset:80\n\t"
            "global_load_async_to_lds_b128 %0, %1, off offset:96\n\t"
            "global_load_async_to_lds_b128 %0, %1, off offset:112"
            :
            : "v"(la), "v"(gk)
            : "memory");
    };

    // ---- Q fragments, scale = 1/sqrt(64) * log2(e) folded in ----
    const float cs = 0.18033688011112042f;
    AFrag aq[2];
    {
        const int t = h * S_ + (q0 + l15);
        const _Float16* qrow =
            qkv + (size_t)(b * S_ + (t >> 4)) * QKVW + (t & 15) * HD + 16 * hi;
#pragma unroll
        for (int dc = 0; dc < 2; ++dc) {
            aq[dc].h[0] = *(const v8h*)(qrow + 32 * dc);
            aq[dc].h[1] = *(const v8h*)(qrow + 32 * dc + 8);
        }
#pragma unroll
        for (int dc = 0; dc < 2; ++dc)
#pragma unroll
            for (int e2 = 0; e2 < 16; ++e2)
                aq[dc].e[e2] = (_Float16)((float)aq[dc].e[e2] * cs);
    }

    v8f o[4] = {};
    float mrun = -__builtin_inff();   // stats for row q0 + l15 (replicated
    float lrun = 0.f;                 // across the two half-waves)

    const int jend = q0 + 16;
    issueK(0, 0);                     // prime the async pipeline

    for (int j0 = 0; j0 < jend; j0 += 32) {
        const int buf = (j0 >> 5) & 1;
        const bool more = (j0 + 32) < jend;

        // ---- issue next block's K async copy; prefetch next V rows ----
        if (more) {
            issueK(j0 + 32, buf ^ 1);
            const int tn = hk * S_ + (j0 + 32 + lane);
            const _Float16* nb =
                qkv + (size_t)(b * S_ + (tn >> 2)) * QKVW + (tn & 3) * HD;
            __builtin_prefetch(nb + H_ + GHD, 0, 3);  // V row
        }

        // ---- stage V^T into LDS: lane owns key j0+lane ----
        {
            const int t = hk * S_ + (j0 + lane);
            const _Float16* vrow = qkv + (size_t)(b * S_ + (t >> 2)) * QKVW +
                                   (H_ + GHD) + (t & 3) * HD;
#pragma unroll
            for (int d8 = 0; d8 < 8; ++d8) {
                v8h vv = *(const v8h*)(vrow + d8 * 8);
#pragma unroll
                for (int i = 0; i < 8; ++i) vT[d8 * 8 + i][lane] = vv[i];
            }
        }

        // ---- wait for current K tile (next tile may stay in flight) ----
        if (more) asm volatile("s_wait_asynccnt 0x8" ::: "memory");
        else      asm volatile("s_wait_asynccnt 0x0" ::: "memory");

        // ---- K fragments from LDS, then 4 score WMMAs ----
        AFrag kb[2][2];
#pragma unroll
        for (int t = 0; t < 2; ++t)
#pragma unroll
            for (int dc = 0; dc < 2; ++dc) {
                kb[t][dc].h[0] =
                    *(const v8h*)&kS[buf][16 * t + l15][32 * dc + 16 * hi];
                kb[t][dc].h[1] =
                    *(const v8h*)&kS[buf][16 * t + l15][32 * dc + 16 * hi + 8];
            }
        v8f sc[2] = {};
#pragma unroll
        for (int t = 0; t < 2; ++t)
#pragma unroll
            for (int dc = 0; dc < 2; ++dc)
                sc[t] = __builtin_amdgcn_wmma_f32_16x16x32_f16(
                    false, aq[dc].v, false, kb[t][dc].v, (short)0, sc[t],
                    false, false);

        // ---- scatter scores to LDS (D layout: row r+8*hi, col 16t+l15) ----
#pragma unroll
        for (int t = 0; t < 2; ++t)
#pragma unroll
            for (int r = 0; r < 8; ++r)
                sS[r + 8 * hi][16 * t + l15] = sc[t][r];
        __syncthreads();

        // ---- row phase: lane reduces row q0+l15, keys j0+16*hi..+15 ----
        {
            const int myrow = q0 + l15;
            const int kbase = j0 + 16 * hi;
            v4f rv[4];
#pragma unroll
            for (int j = 0; j < 4; ++j)
                rv[j] = *(const v4f*)&sS[l15][16 * hi + 4 * j];
            float vals[16];
#pragma unroll
            for (int i = 0; i < 16; ++i) {
                const float x = rv[i >> 2][i & 3];
                vals[i] = (kbase + i <= myrow) ? x : -__builtin_inff();
            }
            float pmax = vals[0];
#pragma unroll
            for (int i = 1; i < 16; ++i) pmax = fmaxf(pmax, vals[i]);
            pmax = fmaxf(pmax, __shfl_xor(pmax, 16, 32));
            const float mnew  = fmaxf(mrun, pmax);
            const float alpha = exp2f(mrun - mnew);
            float psum = 0.f;
            v8h ph[2];
#pragma unroll
            for (int i = 0; i < 16; ++i) {
                const float p = exp2f(vals[i] - mnew);
                psum += p;
                ph[i >> 3][i & 7] = (_Float16)p;
            }
            psum += __shfl_xor(psum, 16, 32);
            lrun = lrun * alpha + psum;
            mrun = mnew;
            *(v8h*)&pT[l15][16 * hi]     = ph[0];
            *(v8h*)&pT[l15][16 * hi + 8] = ph[1];
            if (hi == 0) bcast[l15] = alpha;
        }
        __syncthreads();

        // ---- rescale O by per-row alpha (rows r + 8*hi) ----
        {
            const v4f a0 = *(const v4f*)&bcast[8 * hi];
            const v4f a1 = *(const v4f*)&bcast[8 * hi + 4];
#pragma unroll
            for (int t = 0; t < 4; ++t)
#pragma unroll
                for (int r = 0; r < 8; ++r)
                    o[t][r] *= (r < 4) ? a0[r & 3] : a1[r & 3];
        }

        // ---- P A-fragment + V B-fragments (all vector LDS loads) ----
        AFrag pa;
        pa.h[0] = *(const v8h*)&pT[l15][16 * hi];
        pa.h[1] = *(const v8h*)&pT[l15][16 * hi + 8];
#pragma unroll
        for (int dt = 0; dt < 4; ++dt) {
            AFrag vb;
            vb.h[0] = *(const v8h*)&vT[dt * 16 + l15][16 * hi];
            vb.h[1] = *(const v8h*)&vT[dt * 16 + l15][16 * hi + 8];
            o[dt] = __builtin_amdgcn_wmma_f32_16x16x32_f16(
                false, pa.v, false, vb.v, (short)0, o[dt], false, false);
        }
        __syncthreads();
    }

    // ---- final normalization: broadcast 1/l, scale, store y^T ----
    if (hi == 0) bcast[l15] = 1.f / lrun;
    __syncthreads();
    const v4f i0 = *(const v4f*)&bcast[8 * hi];
    const v4f i1 = *(const v4f*)&bcast[8 * hi + 4];
#pragma unroll
    for (int dt = 0; dt < 4; ++dt)
#pragma unroll
        for (int r = 0; r < 8; ++r) {
            const float val = o[dt][r] * ((r < 4) ? i0[r & 3] : i1[r & 3]);
            yt[(size_t)(b * S_ + q0 + r + 8 * hi) * H_ + h * HD + dt * 16 +
               l15] = (_Float16)val;
        }
}

// ---------------------------------------------------------------------------
extern "C" void kernel_launch(void* const* d_in, const int* in_sizes, int n_in,
                              void* d_out, int out_size, void* d_ws,
                              size_t ws_size, hipStream_t stream) {
    (void)in_sizes; (void)n_in; (void)out_size; (void)ws_size;
    const float* x      = (const float*)d_in[0];  // [B,S,H]
    const float* w_attn = (const float*)d_in[1];  // [1536,1024]
    const float* w_proj = (const float*)d_in[2];  // [1024,1024]
    float* out = (float*)d_out;                   // [B,S,H] f32

    char* ws = (char*)d_ws;
    constexpr size_t XH_B  = (size_t)MROWS * H_ * 2;
    constexpr size_t WAH_B = (size_t)QKVW * H_ * 2;
    constexpr size_t WPH_B = (size_t)H_ * H_ * 2;
    constexpr size_t QKV_B = (size_t)MROWS * QKVW * 2;
    _Float16* xh   = (_Float16*)(ws);
    _Float16* wah  = (_Float16*)(ws + XH_B);
    _Float16* wph  = (_Float16*)(ws + XH_B + WAH_B);
    _Float16* qkvh = (_Float16*)(ws + XH_B + WAH_B + WPH_B);
    _Float16* yth  = (_Float16*)(ws + XH_B + WAH_B + WPH_B + QKV_B);

    // 1) convert inputs to f16
    cvt_f32_f16<<<2048, 256, 0, stream>>>(x, xh, MROWS * H_);
    cvt_f32_f16<<<1024, 256, 0, stream>>>(w_attn, wah, QKVW * H_);
    cvt_f32_f16<<<1024, 256, 0, stream>>>(w_proj, wph, H_ * H_);

    // 2) QKV projection: [4096,1536] = xh @ wah^T  (f16 out)
    {
        const int waves = (MROWS / 16) * (QKVW / 64);   // 6144
        gemm_wmma_f16<_Float16, H_><<<waves / 4, 128, 0, stream>>>(
            xh, wah, qkvh, QKVW);
    }

    // 3) flash attention (one wave per b, head, 16-query tile)
    attn_flash_wmma<<<dim3(S_ / 16, NQ, B_), 32, 0, stream>>>(qkvh, yth);

    // 4) output projection: out[4096,1024] = yth @ wph^T  (f32 out)
    {
        const int waves = (MROWS / 16) * (H_ / 64);     // 4096
        gemm_wmma_f16<float, H_><<<waves / 4, 128, 0, stream>>>(
            yth, wph, out, H_);
    }
}
